// MultiScaleDecoderRetention_3161095930391
// MI455X (gfx1250) — matrix operand order repaired and checked by reading, not compiled, not run on hardware
//
#include <hip/hip_runtime.h>
#include <hip/hip_bf16.h>

typedef __attribute__((ext_vector_type(16))) _Float16 v16h;
typedef __attribute__((ext_vector_type(8)))  float    v8f;
typedef unsigned int uint32x4 __attribute__((ext_vector_type(4)));
typedef int          int32x4  __attribute__((ext_vector_type(4)));
typedef int          int32x8  __attribute__((ext_vector_type(8)));

#define HEADS   8
#define HIDDEN  1024
#define HEADDIM 128
#define SLEN    2048
#define BATCH   4
#define MROWS   (BATCH * SLEN)

// epilogue modes
#define EP_Q    0
#define EP_K    1
#define EP_V    2
#define EP_GATE 3
#define EP_OUT  4

// ---------------------------------------------------------------------------
// Tensor Data Mover support (gfx1250). 5-arg builtin on ROCm 7.2 clang-22;
// 6-arg on the therock toolchain (detected via its TDM header).
// ---------------------------------------------------------------------------
#if defined(__has_builtin)
#  if __has_builtin(__builtin_amdgcn_tensor_load_to_lds) && \
      __has_builtin(__builtin_amdgcn_s_wait_tensorcnt)
#    define HAVE_TDM 1
#  endif
#endif
#ifndef HAVE_TDM
#  define HAVE_TDM 0
#endif
#if __has_include(<hip/amd_detail/amd_gfx1250_TDM.h>)
#  define TDM_SIX_ARGS 1
#else
#  define TDM_SIX_ARGS 0
#endif

#if HAVE_TDM
template <typename T>
__device__ __forceinline__ unsigned lds_off_u32(const T* p) {
  return (unsigned)(unsigned long long)(__attribute__((address_space(3))) const T*)p;
}

// 2D f16 tile load: tile_d0 elems/row, tile_d1 rows, global row stride in elems,
// LDS destination padded by (pad_amt) DWORDs every 2^(pad_int+1) DWORDs.
__device__ __forceinline__ void tdm_load_2d_f16(const void* gsrc, unsigned lds_off,
                                                unsigned tile_d0, unsigned tile_d1,
                                                unsigned tensor_d0, unsigned tensor_d1,
                                                unsigned stride_elems,
                                                unsigned pad_int_code,
                                                unsigned pad_amt_code) {
  const unsigned long long ga = (unsigned long long)(uintptr_t)gsrc;
  uint32x4 g0;
  g0[0] = 1u;                                             // count=1, user D#
  g0[1] = lds_off;                                        // lds_addr (bytes)
  g0[2] = (unsigned)(ga & 0xffffffffull);                 // global_addr[31:0]
  g0[3] = (unsigned)((ga >> 32) & 0x01ffffffull)          // global_addr[56:32]
        | (2u << 30);                                     // type = 2 (image)
  int32x8 g1;
  g1[0] = (int)((1u << 16)                                // data_size = 2 bytes
              | (1u << 20)                                // pad_enable
              | (pad_int_code << 22) | (pad_amt_code << 25));
  g1[1] = (int)((tensor_d0 & 0xffffu) << 16);             // tensor_dim0[15:0]
  g1[2] = (int)((tensor_d0 >> 16) | ((tensor_d1 & 0xffffu) << 16));
  g1[3] = (int)((tensor_d1 >> 16) | (tile_d0 << 16));     // tile_dim0
  g1[4] = (int)(tile_d1 & 0xffffu);                       // tile_dim1 (tile_dim2=0)
  g1[5] = (int)stride_elems;                              // tensor_dim0_stride[31:0]
  g1[6] = 0;                                              // stride hi / dim1_stride lo
  g1[7] = 0;
  int32x4 gz4 = {0, 0, 0, 0};
#if TDM_SIX_ARGS
  int32x8 gz8 = {0, 0, 0, 0, 0, 0, 0, 0};
  __builtin_amdgcn_tensor_load_to_lds(g0, g1, gz4, gz4, gz8, 0);
#else
  __builtin_amdgcn_tensor_load_to_lds(g0, g1, gz4, gz4, 0);
#endif
}
#endif  // HAVE_TDM

// ---------------------------------------------------------------------------
// WMMA fragment loaders from LDS (row-major [idx][k] storage).
// A 16x32 f16: lane&15 = M row; low half-lanes K=0..7,16..23, high K=8..15,24..31.
// B 32x16 f16: lane&15 = N col; low half-lanes K=0..7,16..23, high K=8..15,24..31.
// ---------------------------------------------------------------------------
__device__ __forceinline__ v16h ld_frag_a(const _Float16* p, int stride, int idx0, int k0) {
  const int lane = threadIdx.x & 31;
  const _Float16* r = p + (size_t)(idx0 + (lane & 15)) * stride + k0 + ((lane >> 4) << 3);
  v16h f;
#pragma unroll
  for (int i = 0; i < 8; ++i) f[i] = r[i];
#pragma unroll
  for (int i = 0; i < 8; ++i) f[8 + i] = r[16 + i];
  return f;
}

__device__ __forceinline__ v16h ld_frag_b(const _Float16* p, int stride, int idx0, int k0) {
  const int lane = threadIdx.x & 31;
  const _Float16* r = p + (size_t)(idx0 + (lane & 15)) * stride + k0 + ((lane >> 4) << 3);
  v16h f;
#pragma unroll
  for (int i = 0; i < 8; ++i) f[i] = r[i];
#pragma unroll
  for (int i = 0; i < 8; ++i) f[8 + i] = r[16 + i];
  return f;
}

// ---------------------------------------------------------------------------
// A-tile staging: f32 source converts to f16 manually; f16 source goes
// through the Tensor Data Mover (pad 16 DW + 4 DW matches the [64][40] LDS).
// ---------------------------------------------------------------------------
__device__ __forceinline__ void stage_A(const float* A, _Float16 (*Alds)[40],
                                        int m0, int k0, int tid) {
#pragma unroll
  for (int v = 0; v < 2; ++v) {
    int fi = tid + v * 256;              // 512 float4 = 64x32
    int r = fi >> 3, c4 = (fi & 7) << 2;
    const float4 q = *(const float4*)(A + (size_t)(m0 + r) * HIDDEN + k0 + c4);
    Alds[r][c4 + 0] = (_Float16)q.x; Alds[r][c4 + 1] = (_Float16)q.y;
    Alds[r][c4 + 2] = (_Float16)q.z; Alds[r][c4 + 3] = (_Float16)q.w;
  }
}

__device__ __forceinline__ void stage_A(const _Float16* A, _Float16 (*Alds)[40],
                                        int m0, int k0, int tid) {
#if HAVE_TDM
  if ((tid >> 5) == 0) {
    // 64 rows x 32 f16 (64B rows), LDS rows padded to 80B: interval 16 DW, pad 4 DW
    tdm_load_2d_f16(A + (size_t)m0 * HIDDEN + k0, lds_off_u32(&Alds[0][0]),
                    32, 64, 32, (unsigned)(MROWS - m0), HIDDEN,
                    /*pad_int(16DW)=*/3, /*pad_amt(4DW)=*/3);
    __builtin_amdgcn_s_wait_tensorcnt(0);
  }
#else
  int r = tid >> 2, c8 = (tid & 3) << 3;  // 256 * 8 f16 = 64x32
  *(uint4*)&Alds[r][c8] = *(const uint4*)(A + (size_t)(m0 + r) * HIDDEN + k0 + c8);
#endif
}

// ---------------------------------------------------------------------------
// Generic 64x64-tile GEMM:  C[M=8192, N=1024] = A[.,1024] * W[1024,1024]
// headed=1 -> W laid out [H][1024][128] (per-head projection weights).
// Fused epilogues: xPos(Q), xPos^-1(K), f16 cast (V), SiLU (gate), f32 (out).
// ---------------------------------------------------------------------------
template <typename TA>
__global__ __launch_bounds__(256) void gemm_ep(const TA* __restrict__ A,
                                               const float* __restrict__ W,
                                               void* __restrict__ Out,
                                               int mode, int headed) {
  __shared__ _Float16 Alds[64][40];
  __shared__ _Float16 Blds[64][40];   // stored transposed: [n][k]

  const int tid  = threadIdx.x;
  const int lane = tid & 31;
  const int w    = tid >> 5;
  const int mw   = w & 3;      // 4 row groups of 16
  const int nw   = w >> 2;     // 2 col groups of 32
  const int m0   = blockIdx.y * 64;
  const int n0   = blockIdx.x * 64;

  v8f acc0 = {}; v8f acc1 = {};

  for (int k0 = 0; k0 < HIDDEN; k0 += 32) {
    stage_A(A, Alds, m0, k0, tid);
#pragma unroll
    for (int v = 0; v < 2; ++v) {
      int fi = tid + v * 256;               // 512 float4 = 32x64
      int kk = fi >> 4, c4 = (fi & 15) << 2;
      int n = n0 + c4;
      size_t widx = headed
          ? ((size_t)(n >> 7) * HIDDEN + (k0 + kk)) * HEADDIM + (n & 127)
          : (size_t)(k0 + kk) * HIDDEN + n;
      const float4 q = *(const float4*)(W + widx);
      Blds[c4 + 0][kk] = (_Float16)q.x; Blds[c4 + 1][kk] = (_Float16)q.y;
      Blds[c4 + 2][kk] = (_Float16)q.z; Blds[c4 + 3][kk] = (_Float16)q.w;
    }
    if (k0 + 32 < HIDDEN)
      __builtin_prefetch((const void*)(W + (size_t)(k0 + 32 + (tid >> 3)) * HIDDEN + n0), 0, 1);
    __syncthreads();

    v16h af  = ld_frag_a(&Alds[0][0], 40, mw * 16, 0);
    v16h bf0 = ld_frag_b(&Blds[0][0], 40, nw * 32, 0);
    v16h bf1 = ld_frag_b(&Blds[0][0], 40, nw * 32 + 16, 0);
    acc0 = __builtin_amdgcn_wmma_f32_16x16x32_f16(false, af, false, bf0, (short)0, acc0, false, false);
    acc1 = __builtin_amdgcn_wmma_f32_16x16x32_f16(false, af, false, bf1, (short)0, acc1, false, false);
    __syncthreads();
  }

  // ---------------- epilogue ----------------
  const int hi = lane >> 4;
#pragma unroll
  for (int a = 0; a < 2; ++a) {
    v8f ac = a ? acc1 : acc0;
    const int n = n0 + nw * 32 + a * 16 + (lane & 15);
#pragma unroll
    for (int i = 0; i < 8; ++i) {
      const int m = m0 + mw * 16 + 8 * hi + i;
      float y = ac[i];
      if (mode == EP_Q || mode == EP_K) {
        // xPos rotary with length-extrapolatable scale
        const int s = m & (SLEN - 1);
        const int e = n & (HEADDIM - 1);
        const int j = e >> 1;
        const float inv_freq = exp2f(-(float)j * (13.287712379549449f / 64.0f)); // 10000^(-j/64)
        const float ang = (float)s * inv_freq;
        const float sn = __sinf(ang), cs = __cosf(ang);
        const float base = ((float)(2 * j) + 51.2f) * (1.0f / 179.2f);
        const float pw = (float)s * (1.0f / 512.0f);
        const float sc = exp2f(((mode == EP_Q) ? pw : -pw) * log2f(base));
        const float c_s = cs * sc, s_s = sn * sc;
        const float part = __shfl_xor(y, 1, 32);          // partner within rotary pair
        y = (e & 1) ? (y * c_s + part * s_s) : (y * c_s - part * s_s);
      } else if (mode == EP_GATE) {
        y = y / (1.0f + __expf(-y));                      // SiLU
      }
      if (mode == EP_OUT)
        ((float*)Out)[(size_t)m * HIDDEN + n] = y;
      else
        ((_Float16*)Out)[(size_t)m * HIDDEN + n] = (_Float16)y;
    }
  }
}

// ---------------------------------------------------------------------------
// Retention core: per (b, h, 64-row query tile)
//   ret = (Q Kᵀ) .* γ^(s-t) (causal)   ;   Y = ret @ V
// fused with per-head group-norm + gate multiply; writes f16 gated output.
// Q and K tiles stream in via the Tensor Data Mover (wave 0 issues the DMA,
// waits TENSORcnt, then the workgroup barrier publishes the LDS tile).
// ---------------------------------------------------------------------------
__global__ __launch_bounds__(256) void retention_kernel(
    const _Float16* __restrict__ Qh, const _Float16* __restrict__ Kh,
    const _Float16* __restrict__ Vh, const _Float16* __restrict__ Gate,
    const float* __restrict__ gn_w, const float* __restrict__ gn_b,
    _Float16* __restrict__ Yg) {
  __shared__ _Float16 Qs[64][136];     // Q tile, row-major [s][k] (256B + 16B pad rows)
  __shared__ _Float16 Ks[32][136];     // K tile, row-major [t][k] (== Kᵀ col-major)
  __shared__ _Float16 Vt[128][40];     // V tile transposed [n][t]
  __shared__ _Float16 Rs[8][16][40];   // per-wave decayed ret scratch [m][t]
  __shared__ float redS[2][64];
  __shared__ float redQ[2][64];

  const int tid  = threadIdx.x;
  const int lane = tid & 31;
  const int w    = tid >> 5;
  const int mw   = w & 3;              // 16-row group inside 64-row s tile
  const int nw   = w >> 2;             // 64-col half of the 128-wide head
  const int s0   = blockIdx.x * 64;
  const int h    = blockIdx.y;
  const int b    = blockIdx.z;
  const size_t rowbase = (size_t)b * SLEN * HIDDEN + (size_t)h * HEADDIM;

  // gamma_h = 1 - exp(linspace(ln 1/32, ln 1/512, 8))
  const float lg_lo = -3.4657359027997265f;   // ln(1/32)
  const float lg_hi = -6.238324625039508f;    // ln(1/512)
  const float gamma = 1.0f - __expf(lg_lo + (lg_hi - lg_lo) * ((float)h / 7.0f));
  const float lg2g  = log2f(gamma);

  // stage Q tile 64x128 once (TDM: rows 256B, LDS pad 64DW interval + 4DW)
#if HAVE_TDM
  if (w == 0) {
    tdm_load_2d_f16(Qh + rowbase + (size_t)s0 * HIDDEN, lds_off_u32(&Qs[0][0]),
                    128, 64, 128, (unsigned)(SLEN - s0), HIDDEN,
                    /*pad_int(64DW)=*/5, /*pad_amt(4DW)=*/3);
  }
#else
#pragma unroll
  for (int v = 0; v < 4; ++v) {
    int fi = tid + v * 256;            // 1024 x (8 f16)
    int r = fi >> 4, c8 = (fi & 15) << 3;
    *(uint4*)&Qs[r][c8] = *(const uint4*)(Qh + rowbase + (size_t)(s0 + r) * HIDDEN + c8);
  }
#endif

  v8f Y[4] = {};
  const int hi = lane >> 4;
  const int t_end = s0 + 64;           // causal bound for this tile

  for (int t0 = 0; t0 < t_end; t0 += 32) {
    // stage K via TDM; V transposed manually (TDM cannot transpose)
#if HAVE_TDM
    if (w == 0) {
      tdm_load_2d_f16(Kh + rowbase + (size_t)t0 * HIDDEN, lds_off_u32(&Ks[0][0]),
                      128, 32, 128, (unsigned)(SLEN - t0), HIDDEN,
                      /*pad_int(64DW)=*/5, /*pad_amt(4DW)=*/3);
    }
#endif
#pragma unroll
    for (int v = 0; v < 2; ++v) {
      int fi = tid + v * 256;          // 512 x (8 f16) = 32x128
      int r = fi >> 4, c8 = (fi & 15) << 3;
#if !HAVE_TDM
      *(uint4*)&Ks[r][c8] = *(const uint4*)(Kh + rowbase + (size_t)(t0 + r) * HIDDEN + c8);
#endif
      union { uint4 u; _Float16 e[8]; } vv;
      vv.u = *(const uint4*)(Vh + rowbase + (size_t)(t0 + r) * HIDDEN + c8);
#pragma unroll
      for (int j = 0; j < 8; ++j) Vt[c8 + j][r] = vv.e[j];
    }
#if HAVE_TDM
    if (w == 0) __builtin_amdgcn_s_wait_tensorcnt(0);  // covers Q on iter 0 too
#endif
    __syncthreads();

    // --- ret = Q Kᵀ over this 32-wide t chunk (K-dim = 128 = 4 WMMA steps) ---
    v16h qf[4];
#pragma unroll
    for (int kk = 0; kk < 4; ++kk) qf[kk] = ld_frag_a(&Qs[0][0], 136, mw * 16, kk * 32);

#pragma unroll
    for (int a = 0; a < 2; ++a) {
      v8f rr = {};
#pragma unroll
      for (int kk = 0; kk < 4; ++kk) {
        v16h kf = ld_frag_b(&Ks[0][0], 136, a * 16, kk * 32);   // Kᵀ fragment
        rr = __builtin_amdgcn_wmma_f32_16x16x32_f16(false, qf[kk], false, kf, (short)0, rr, false, false);
      }
      // decay + causal mask, relayout into wave-private LDS scratch
      const int tcol = t0 + a * 16 + (lane & 15);
#pragma unroll
      for (int i = 0; i < 8; ++i) {
        const int srow = s0 + mw * 16 + 8 * hi + i;
        const float diff = (float)(srow - tcol);
        const float dfac = (diff >= 0.0f) ? exp2f(diff * lg2g) : 0.0f;
        Rs[w][8 * hi + i][a * 16 + (lane & 15)] = (_Float16)(rr[i] * dfac);
      }
    }

    // --- Y += ret(16x32) @ V(32x64 half) ---
    v16h rf = ld_frag_a(&Rs[w][0][0], 40, 0, 0);
#pragma unroll
    for (int j = 0; j < 4; ++j) {
      v16h vf = ld_frag_b(&Vt[0][0], 40, nw * 64 + j * 16, 0);
      Y[j] = __builtin_amdgcn_wmma_f32_16x16x32_f16(false, rf, false, vf, (short)0, Y[j], false, false);
    }
    __syncthreads();
  }

  // --- group norm over the head dim (128) per row ---
#pragma unroll
  for (int i = 0; i < 8; ++i) {
    float s = 0.0f, q = 0.0f;
#pragma unroll
    for (int j = 0; j < 4; ++j) { const float v = Y[j][i]; s += v; q += v * v; }
#pragma unroll
    for (int d = 1; d < 16; d <<= 1) { s += __shfl_xor(s, d, 32); q += __shfl_xor(q, d, 32); }
    if ((lane & 15) == 0) {
      redS[nw][mw * 16 + 8 * hi + i] = s;
      redQ[nw][mw * 16 + 8 * hi + i] = q;
    }
  }
  __syncthreads();

  // normalize, affine, gate; stage to Qs (reused) for coalesced store
#pragma unroll
  for (int j = 0; j < 4; ++j) {
    const int n = nw * 64 + j * 16 + (lane & 15);
    const float gw = gn_w[h * HEADDIM + n];
    const float gb = gn_b[h * HEADDIM + n];
#pragma unroll
    for (int i = 0; i < 8; ++i) {
      const int rloc = mw * 16 + 8 * hi + i;
      const float mu  = (redS[0][rloc] + redS[1][rloc]) * (1.0f / 128.0f);
      const float ex2 = (redQ[0][rloc] + redQ[1][rloc]) * (1.0f / 128.0f);
      const float inv = rsqrtf(ex2 - mu * mu + 1e-5f);
      const float yn  = (Y[j][i] - mu) * inv * gw + gb;
      const float g   = (float)Gate[rowbase + (size_t)(s0 + rloc) * HIDDEN + n];
      Qs[rloc][n] = (_Float16)(yn * g);
    }
  }
  __syncthreads();

#pragma unroll
  for (int v = 0; v < 4; ++v) {
    int fi = tid + v * 256;
    int r = fi >> 4, c8 = (fi & 15) << 3;
    *(uint4*)(Yg + rowbase + (size_t)(s0 + r) * HIDDEN + c8) = *(const uint4*)&Qs[r][c8];
  }
}

// ---------------------------------------------------------------------------
extern "C" void kernel_launch(void* const* d_in, const int* in_sizes, int n_in,
                              void* d_out, int out_size, void* d_ws, size_t ws_size,
                              hipStream_t stream) {
  const float* X    = (const float*)d_in[0];
  const float* Mem  = (const float*)d_in[1];
  const float* WQ   = (const float*)d_in[2];
  const float* WK   = (const float*)d_in[3];
  const float* WV   = (const float*)d_in[4];
  const float* W_G  = (const float*)d_in[5];
  const float* W_O  = (const float*)d_in[6];
  const float* gn_w = (const float*)d_in[7];
  const float* gn_b = (const float*)d_in[8];
  float* Out = (float*)d_out;

  const size_t elems = (size_t)MROWS * HIDDEN;           // 8,388,608
  char* ws = (char*)d_ws;
  _Float16* Qf = (_Float16*)(ws);
  _Float16* Kf = (_Float16*)(ws + 2 * elems);
  _Float16* Vf = (_Float16*)(ws + 4 * elems);
  _Float16* Gf = (_Float16*)(ws + 6 * elems);
  _Float16* Yg = (_Float16*)(ws + 8 * elems);

  dim3 blk(256);
  dim3 gg(HIDDEN / 64, MROWS / 64);                      // 16 x 128
  gemm_ep<float><<<gg, blk, 0, stream>>>(X,   WQ,  (void*)Qf, EP_Q,    1);
  gemm_ep<float><<<gg, blk, 0, stream>>>(Mem, WK,  (void*)Kf, EP_K,    1);
  gemm_ep<float><<<gg, blk, 0, stream>>>(Mem, WV,  (void*)Vf, EP_V,    1);
  gemm_ep<float><<<gg, blk, 0, stream>>>(X,   W_G, (void*)Gf, EP_GATE, 0);

  dim3 gr(SLEN / 64, HEADS, BATCH);                      // 32 x 8 x 4
  retention_kernel<<<gr, blk, 0, stream>>>(Qf, Kf, Vf, Gf, gn_w, gn_b, Yg);

  gemm_ep<_Float16><<<gg, blk, 0, stream>>>(Yg, W_O, (void*)Out, EP_OUT, 0);
}
